// DoubleStreamBlock_35064113004851
// MI455X (gfx1250) — compile-verified, matrix operand order checked by reading, not run on hardware
//
#include <hip/hip_runtime.h>
#include <stdint.h>

typedef unsigned short u16;
typedef unsigned int   u32;
typedef __attribute__((ext_vector_type(16))) __bf16 v16bf;
typedef __attribute__((ext_vector_type(8)))  float  v8f;
typedef __attribute__((ext_vector_type(4)))  unsigned int u32x4;
typedef __attribute__((ext_vector_type(8)))  int          i32x8;
typedef __attribute__((ext_vector_type(4)))  int          i32x4;

// ---------------------------------------------------------------------------
// helpers
// ---------------------------------------------------------------------------
__device__ __forceinline__ u16 f2bf(float x) {
    __bf16 b = (__bf16)x;                    // native v_cvt on gfx1250 (RTNE)
    return __builtin_bit_cast(u16, b);
}
__device__ __forceinline__ float gelu_tanh(float x) {
    float x3 = x * x * x;
    float t  = tanhf(0.7978845608028654f * (x + 0.044715f * x3));
    return 0.5f * x * (1.0f + t);
}
__device__ __forceinline__ v8f vzero8() {
    v8f z = {0.f, 0.f, 0.f, 0.f, 0.f, 0.f, 0.f, 0.f};
    return z;
}

union FragBF { u32 u[8]; v16bf v; };

__device__ __forceinline__ v8f wmma_bf16(const FragBF& a, const FragBF& b, v8f c) {
#if defined(__AMDGCN__)
    return __builtin_amdgcn_wmma_f32_16x16x32_bf16(false, a.v, false, b.v,
                                                   (short)0, c, false, false);
#else
    (void)a; (void)b;
    return c;
#endif
}

// K index inside a 32-wide K tile for 16-bit A/B fragments (ISA 7.12.2):
// lanes 0-15: VGPR 0-3 -> K=0..7, VGPR 4-7 -> K=16..23
// lanes16-31: VGPR 0-3 -> K=8..15, VGPR 4-7 -> K=24..31
__device__ __forceinline__ int kmap(int r, int g) {
    return ((r & 4) ? 16 : 0) + g * 8 + (r & 3) * 2;
}

// ---------------------------------------------------------------------------
// Tensor Data Mover: 2-D bf16 tile (tile_k x tile_rows) global -> LDS.
// D# per ISA 08_async_tensor §8.3/8.4; groups 2/3 zero (<=2D tensor).
// clang-23 6-arg builtin: (v4u g0, v8i g1, v4i g2, v4i g3, v8i, i32 cpol).
// ---------------------------------------------------------------------------
__device__ __forceinline__ void tdm_load_2d_bf16(u32 lds_off, const void* gaddr,
                                                 u32 tile_k, u32 tile_rows,
                                                 u32 row_stride_elems) {
#if defined(__AMDGCN__)
    unsigned long long ga = (unsigned long long)(uintptr_t)gaddr;
    u32x4 g0;
    g0[0] = 1u;                                        // count=1, user D#
    g0[1] = lds_off;                                   // lds_addr (bytes)
    g0[2] = (u32)(ga & 0xFFFFFFFFu);                   // global_addr[31:0]
    g0[3] = (u32)((ga >> 32) & 0x01FFFFFFu) | (2u << 30); // ga[56:32] | type=2
    i32x8 g1;
    g1[0] = (int)(1u << 16);                           // data_size=1 -> 2 bytes
    g1[1] = (int)((tile_k & 0xFFFFu) << 16);           // tensor_dim0[15:0]
    g1[2] = (int)((tile_k >> 16) | ((tile_rows & 0xFFFFu) << 16)); // td0 hi|td1 lo
    g1[3] = (int)((tile_rows >> 16) | ((tile_k & 0xFFFFu) << 16)); // td1 hi|tile_dim0
    g1[4] = (int)(tile_rows & 0xFFFFu);                // tile_dim1 (tile_dim2=0)
    g1[5] = (int)row_stride_elems;                     // tensor_dim0_stride[31:0]
    g1[6] = 0;                                         // stride hi | dim1_stride lo
    g1[7] = 0;
    i32x4 gz4 = {0, 0, 0, 0};
    i32x8 gz8 = {0, 0, 0, 0, 0, 0, 0, 0};
    __builtin_amdgcn_tensor_load_to_lds(g0, g1, gz4, gz4, gz8, 0);
#else
    (void)lds_off; (void)gaddr; (void)tile_k; (void)tile_rows; (void)row_stride_elems;
#endif
}
__device__ __forceinline__ void wait_tensor0() {
#if defined(__AMDGCN__)
    __builtin_amdgcn_s_wait_tensorcnt(0);
#endif
}

// ---------------------------------------------------------------------------
// 1. modulation: out[j] = sum_i silu(vec[i]) * W[i,j] + b[j]
// ---------------------------------------------------------------------------
__global__ void modulation_kernel(const float* __restrict__ vec,
                                  const float* __restrict__ W,
                                  const float* __restrict__ b,
                                  float* __restrict__ out, int D, int N) {
    __shared__ float sv[3072];
    int tid = threadIdx.x;
    for (int i = tid; i < D; i += 256) {
        float v = vec[i];
        sv[i] = v / (1.0f + __expf(-v));
    }
    __syncthreads();
    int j = blockIdx.x * 256 + tid;
    if (j >= N) return;
    float acc = 0.f;
    for (int i = 0; i < D; ++i)
        acc = fmaf(sv[i], W[(size_t)i * N + j], acc);
    out[j] = acc + b[j];
}

// ---------------------------------------------------------------------------
// 2. layernorm + modulate -> bf16
// ---------------------------------------------------------------------------
__global__ void ln_modulate_kernel(const float* __restrict__ x,
                                   const float* __restrict__ mod,
                                   int shift_off, int scale_off,
                                   u16* __restrict__ y, int D) {
    int row = blockIdx.x;
    int tid = threadIdx.x;
    const float* xr = x + (size_t)row * D;
    float s = 0.f, s2 = 0.f;
    for (int c = tid; c < D; c += 256) {
        float v = xr[c];
        s += v; s2 += v * v;
    }
    __shared__ float rs[8], rs2[8];
    for (int off = 1; off < 32; off <<= 1) {
        s  += __shfl_xor(s, off, 32);
        s2 += __shfl_xor(s2, off, 32);
    }
    if ((tid & 31) == 0) { rs[tid >> 5] = s; rs2[tid >> 5] = s2; }
    __syncthreads();
    s = 0.f; s2 = 0.f;
    #pragma unroll
    for (int i = 0; i < 8; ++i) { s += rs[i]; s2 += rs2[i]; }
    float mu   = s / (float)D;
    float var  = s2 / (float)D - mu * mu;
    float rstd = rsqrtf(var + 1e-6f);
    for (int c = tid; c < D; c += 256) {
        float v = (xr[c] - mu) * rstd;
        v = v * (1.0f + mod[scale_off + c]) + mod[shift_off + c];
        y[(size_t)row * D + c] = f2bf(v);
    }
}

// ---------------------------------------------------------------------------
// 3. WMMA GEMM: C(MxN) = A(MxK,bf16) * W(KxN,f32->bf16) + bias
//    block tile 128x128, 8 waves; each wave 32x64 = 2x4 WMMA 16x16x32 tiles.
//    A tile arrives via Tensor Data Mover (TENSORcnt); B staged + converted.
// ---------------------------------------------------------------------------
__global__ void gemm_bf16_wmma(const u16* __restrict__ A,
                               const float* __restrict__ W,
                               const float* __restrict__ bias,
                               float* __restrict__ outF,
                               u16* __restrict__ outB,
                               int M, int N, int K, int do_gelu) {
    __shared__ u16 lsA[128 * 32];   // [m][k]
    __shared__ u16 lsB[128 * 32];   // [n][k] (transposed: contiguous K pairs)

    const int tid  = threadIdx.x;
    const int w    = tid >> 5;
    const int lane = tid & 31;
    const int g    = lane >> 4;
    const int ln   = lane & 15;
    const int wm   = w & 3;         // 4 M sub-tiles of 32 rows
    const int wn   = w >> 2;        // 2 N sub-tiles of 64 cols
    const int m0   = blockIdx.y * 128;
    const int n0   = blockIdx.x * 128;
    const int kt_n = K >> 5;

    const u32 lsA_off = (u32)(uintptr_t)(&lsA[0]);

    v8f acc[2][4];
    #pragma unroll
    for (int i = 0; i < 2; ++i)
        #pragma unroll
        for (int j = 0; j < 4; ++j) acc[i][j] = vzero8();

    for (int kt = 0; kt < kt_n; ++kt) {
        __syncthreads();
        // A tile 128 rows x 32 bf16 via TDM (wave 0 issues; TDM ignores EXEC)
        if (tid < 32)
            tdm_load_2d_bf16(lsA_off, A + (size_t)m0 * K + kt * 32,
                             32u, 128u, (u32)K);
        // B tile: 32 k-rows x 128 n-cols f32 -> bf16, transposed into LDS
        for (int c = tid; c < 1024; c += 256) {
            int k = c >> 5, seg = c & 31;
            const float4* src =
                (const float4*)(W + (size_t)(kt * 32 + k) * N + n0 + seg * 4);
            float4 f = *src;
            int nb = seg * 4;
            lsB[(nb + 0) * 32 + k] = f2bf(f.x);
            lsB[(nb + 1) * 32 + k] = f2bf(f.y);
            lsB[(nb + 2) * 32 + k] = f2bf(f.z);
            lsB[(nb + 3) * 32 + k] = f2bf(f.w);
        }
        if (kt + 1 < kt_n)   // pull next W tile toward L2 (global_prefetch_b8)
            __builtin_prefetch(W + (size_t)((kt + 1) * 32 + (tid & 31)) * N + n0, 0, 1);
        wait_tensor0();      // wave0: TDM done; others: no-op (TENSORcnt==0)
        __syncthreads();

        FragBF a[2], b[4];
        #pragma unroll
        for (int i = 0; i < 2; ++i) {
            int mrow = wm * 32 + i * 16 + ln;
            #pragma unroll
            for (int r = 0; r < 8; ++r)
                a[i].u[r] = *(const u32*)&lsA[mrow * 32 + kmap(r, g)];
        }
        #pragma unroll
        for (int j = 0; j < 4; ++j) {
            int ncol = wn * 64 + j * 16 + ln;
            #pragma unroll
            for (int r = 0; r < 8; ++r)
                b[j].u[r] = *(const u32*)&lsB[ncol * 32 + kmap(r, g)];
        }
        #pragma unroll
        for (int i = 0; i < 2; ++i)
            #pragma unroll
            for (int j = 0; j < 4; ++j)
                acc[i][j] = wmma_bf16(a[i], b[j], acc[i][j]);
    }

    // epilogue
    #pragma unroll
    for (int i = 0; i < 2; ++i) {
        #pragma unroll
        for (int j = 0; j < 4; ++j) {
            int col = n0 + wn * 64 + j * 16 + ln;
            float bcol = bias[col];
            #pragma unroll
            for (int v = 0; v < 8; ++v) {
                int row = m0 + wm * 32 + i * 16 + g * 8 + v;
                float val = acc[i][j][v] + bcol;
                if (do_gelu) val = gelu_tanh(val);
                if (outB) outB[(size_t)row * N + col] = f2bf(val);
                else      outF[(size_t)row * N + col] = val;
            }
        }
    }
}

// ---------------------------------------------------------------------------
// 4. split qkv, rms-norm q/k per head, fold softmax scale into q, -> bf16
//    Q/K/V layout: [H][Ltot][128], cond tokens first (loff).
// ---------------------------------------------------------------------------
__global__ void qkv_rms_kernel(const float* __restrict__ qkv,
                               const float* __restrict__ qs,
                               const float* __restrict__ ks,
                               u16* __restrict__ Qb, u16* __restrict__ Kb,
                               u16* __restrict__ Vb, int loff, int Ltot) {
    const int l = blockIdx.x;
    const int d = threadIdx.x;                 // 128 threads
    const float sm_scale = 0.08838834764831845f;   // 1/sqrt(128)
    __shared__ float red[8];
    for (int h = 0; h < 24; ++h) {
        float q = qkv[(size_t)l * 9216 + h * 128 + d];
        float k = qkv[(size_t)l * 9216 + 3072 + h * 128 + d];
        float v = qkv[(size_t)l * 9216 + 6144 + h * 128 + d];
        float sq = q * q, sk = k * k;
        for (int off = 1; off < 32; off <<= 1) {
            sq += __shfl_xor(sq, off, 32);
            sk += __shfl_xor(sk, off, 32);
        }
        int wv = d >> 5;
        if ((d & 31) == 0) { red[wv] = sq; red[4 + wv] = sk; }
        __syncthreads();
        float tq = red[0] + red[1] + red[2] + red[3];
        float tk = red[4] + red[5] + red[6] + red[7];
        __syncthreads();
        float rq = rsqrtf(tq / 128.f + 1e-6f);
        float rk = rsqrtf(tk / 128.f + 1e-6f);
        size_t o = ((size_t)h * Ltot + loff + l) * 128 + d;
        Qb[o] = f2bf(q * rq * qs[d] * sm_scale);
        Kb[o] = f2bf(k * rk * ks[d]);
        Vb[o] = f2bf(v);
    }
}

// ---------------------------------------------------------------------------
// 5. flash attention, bf16 WMMA. grid = (Ltot/128, H), 256 threads (8 waves),
//    each wave owns 16 q rows; streams 32-key blocks.
// ---------------------------------------------------------------------------
__global__ void attn_kernel(const u16* __restrict__ Q, const u16* __restrict__ K,
                            const u16* __restrict__ V, u16* __restrict__ Out,
                            int Ltot) {
    const int h   = blockIdx.y;
    const int q0  = blockIdx.x * 128;
    const int tid = threadIdx.x;
    const int w = tid >> 5, lane = tid & 31, g = lane >> 4, ln = lane & 15;

    const u16* Qh = Q + (size_t)h * Ltot * 128;
    const u16* Kh = K + (size_t)h * Ltot * 128;
    const u16* Vh = V + (size_t)h * Ltot * 128;

    __shared__ u16 lsVt[128 * 32];     // [d][key]
    __shared__ u16 lsP[8][16 * 32];    // per-wave P tile (rows x 32 keys)

    // resident Q A-frags for this wave's 16 rows: d = 0..127 in 4 K-tiles
    FragBF aQ[4];
    {
        int qrow = q0 + w * 16 + ln;
        #pragma unroll
        for (int kk = 0; kk < 4; ++kk)
            #pragma unroll
            for (int r = 0; r < 8; ++r) {
                int d = kk * 32 + kmap(r, g);
                aQ[kk].u[r] = *(const u32*)&Qh[(size_t)qrow * 128 + d];
            }
    }

    float mrow[8], lrow[8];
    v8f oacc[8];
    #pragma unroll
    for (int v = 0; v < 8; ++v) { mrow[v] = -1e30f; lrow[v] = 0.f; }
    #pragma unroll
    for (int nt = 0; nt < 8; ++nt) oacc[nt] = vzero8();

    for (int kb = 0; kb < Ltot; kb += 32) {
        __syncthreads();
        // stage V^T tile: 32 keys x 128 d  ->  lsVt[d][key]
        for (int c = tid; c < 32 * 64; c += 256) {
            int key = c >> 6;
            int dp  = c & 63;
            u32 val = *(const u32*)&Vh[(size_t)(kb + key) * 128 + dp * 2];
            lsVt[(dp * 2 + 0) * 32 + key] = (u16)(val & 0xFFFFu);
            lsVt[(dp * 2 + 1) * 32 + key] = (u16)(val >> 16);
        }
        __syncthreads();

        // scores S = Q K^T for 16 rows x 32 keys
        v8f s[2];
        s[0] = vzero8(); s[1] = vzero8();
        #pragma unroll
        for (int j = 0; j < 2; ++j) {
            int keyc = kb + j * 16 + ln;
            #pragma unroll
            for (int kk = 0; kk < 4; ++kk) {
                FragBF bK;
                #pragma unroll
                for (int r = 0; r < 8; ++r) {
                    int d = kk * 32 + kmap(r, g);
                    bK.u[r] = *(const u32*)&Kh[(size_t)keyc * 128 + d];
                }
                s[j] = wmma_bf16(aQ[kk], bK, s[j]);
            }
        }

        // online softmax (rows live in 16-lane half-wave groups)
        #pragma unroll
        for (int v = 0; v < 8; ++v) {
            float mv = fmaxf(s[0][v], s[1][v]);
            for (int off = 1; off < 16; off <<= 1)
                mv = fmaxf(mv, __shfl_xor(mv, off, 32));
            float mnew = fmaxf(mrow[v], mv);
            float corr = __expf(mrow[v] - mnew);
            float p0 = __expf(s[0][v] - mnew);
            float p1 = __expf(s[1][v] - mnew);
            s[0][v] = p0; s[1][v] = p1;
            float rsum = p0 + p1;
            for (int off = 1; off < 16; off <<= 1)
                rsum += __shfl_xor(rsum, off, 32);
            lrow[v] = lrow[v] * corr + rsum;
            mrow[v] = mnew;
            #pragma unroll
            for (int nt = 0; nt < 8; ++nt) oacc[nt][v] *= corr;
        }

        // P (C layout) -> LDS -> A-frag layout
        #pragma unroll
        for (int v = 0; v < 8; ++v) {
            int row = g * 8 + v;
            lsP[w][row * 32 + ln]      = f2bf(s[0][v]);
            lsP[w][row * 32 + 16 + ln] = f2bf(s[1][v]);
        }
        __syncthreads();
        FragBF aP;
        #pragma unroll
        for (int r = 0; r < 8; ++r)
            aP.u[r] = *(const u32*)&lsP[w][ln * 32 + kmap(r, g)];

        // O += P V   (K-dim = 32 keys, 8 N-tiles over d)
        #pragma unroll
        for (int nt = 0; nt < 8; ++nt) {
            FragBF bV;
            int d = nt * 16 + ln;
            #pragma unroll
            for (int r = 0; r < 8; ++r)
                bV.u[r] = *(const u32*)&lsVt[d * 32 + kmap(r, g)];
            oacc[nt] = wmma_bf16(aP, bV, oacc[nt]);
        }
    }

    // write O / l  ->  attn buffer (token, h*128+d) bf16
    #pragma unroll
    for (int v = 0; v < 8; ++v) {
        float rl = 1.0f / lrow[v];
        int row = q0 + w * 16 + g * 8 + v;
        #pragma unroll
        for (int nt = 0; nt < 8; ++nt) {
            int d = nt * 16 + ln;
            Out[(size_t)row * 3072 + h * 128 + d] = f2bf(oacc[nt][v] * rl);
        }
    }
}

// ---------------------------------------------------------------------------
// 6. residual gate: out = x + gate[col] * y
// ---------------------------------------------------------------------------
__global__ void residual_gate_kernel(const float* __restrict__ x,
                                     const float* __restrict__ y,
                                     const float* __restrict__ mod, int gate_off,
                                     float* __restrict__ out, int D, long long n) {
    long long i = (long long)blockIdx.x * 256 + threadIdx.x;
    if (i >= n) return;
    int col = (int)(i - (i / D) * D);
    out[i] = x[i] + mod[gate_off + col] * y[i];
}

// ---------------------------------------------------------------------------
// launcher
// ---------------------------------------------------------------------------
static inline char* align_up(char* p, size_t a) {
    return (char*)(((uintptr_t)p + a - 1) & ~(uintptr_t)(a - 1));
}

extern "C" void kernel_launch(void* const* d_in, const int* in_sizes, int n_in,
                              void* d_out, int out_size, void* d_ws, size_t ws_size,
                              hipStream_t stream) {
    (void)in_sizes; (void)n_in; (void)out_size; (void)ws_size;

    const int D = 3072, Dm = 12288, N3 = 9216, N6 = 18432;
    const int Li = 2048, Lc = 256, Lt = 2304, H = 24;

    const float* img         = (const float*)d_in[0];
    const float* cond        = (const float*)d_in[1];
    const float* vec         = (const float*)d_in[2];
    const float* img_mod_w   = (const float*)d_in[3];
    const float* img_mod_b   = (const float*)d_in[4];
    const float* cond_mod_w  = (const float*)d_in[5];
    const float* cond_mod_b  = (const float*)d_in[6];
    const float* img_qkv_w   = (const float*)d_in[7];
    const float* img_qkv_b   = (const float*)d_in[8];
    const float* img_q_scale = (const float*)d_in[9];
    const float* img_k_scale = (const float*)d_in[10];
    const float* img_proj_w  = (const float*)d_in[11];
    const float* img_proj_b  = (const float*)d_in[12];
    const float* img_mlp_w1  = (const float*)d_in[13];
    const float* img_mlp_b1  = (const float*)d_in[14];
    const float* img_mlp_w2  = (const float*)d_in[15];
    const float* img_mlp_b2  = (const float*)d_in[16];
    const float* cond_qkv_w  = (const float*)d_in[17];
    const float* cond_qkv_b  = (const float*)d_in[18];
    const float* cond_q_scale= (const float*)d_in[19];
    const float* cond_k_scale= (const float*)d_in[20];
    const float* cond_proj_w = (const float*)d_in[21];
    const float* cond_proj_b = (const float*)d_in[22];
    const float* cond_mlp_w1 = (const float*)d_in[23];
    const float* cond_mlp_b1 = (const float*)d_in[24];
    const float* cond_mlp_w2 = (const float*)d_in[25];
    const float* cond_mlp_b2 = (const float*)d_in[26];

    float* out_img  = (float*)d_out;
    float* out_cond = out_img + (size_t)Li * D;

    char* p = (char*)d_ws;
    auto alloc = [&](size_t bytes) -> void* {
        p = align_up(p, 256);
        void* r = (void*)p;
        p += bytes;
        return r;
    };

    float* modI   = (float*)alloc((size_t)N6 * 4);
    float* modC   = (float*)alloc((size_t)N6 * 4);
    u16*   xmI    = (u16*)  alloc((size_t)Li * D * 2);
    u16*   xmC    = (u16*)  alloc((size_t)Lc * D * 2);
    float* qkvI   = (float*)alloc((size_t)Li * N3 * 4);
    float* qkvC   = (float*)alloc((size_t)Lc * N3 * 4);
    u16*   Qb     = (u16*)  alloc((size_t)H * Lt * 128 * 2);
    u16*   Kb     = (u16*)  alloc((size_t)H * Lt * 128 * 2);
    u16*   Vb     = (u16*)  alloc((size_t)H * Lt * 128 * 2);
    u16*   attnB  = (u16*)  alloc((size_t)Lt * D * 2);
    float* projI  = (float*)alloc((size_t)Li * D * 4);
    float* projC  = (float*)alloc((size_t)Lc * D * 4);
    float* img1   = (float*)alloc((size_t)Li * D * 4);
    float* cond1  = (float*)alloc((size_t)Lc * D * 4);
    u16*   xm2I   = (u16*)  alloc((size_t)Li * D * 2);
    u16*   xm2C   = (u16*)  alloc((size_t)Lc * D * 2);
    u16*   hI     = (u16*)  alloc((size_t)Li * Dm * 2);
    u16*   hC     = (u16*)  alloc((size_t)Lc * Dm * 2);
    float* mlp2I  = (float*)alloc((size_t)Li * D * 4);
    float* mlp2C  = (float*)alloc((size_t)Lc * D * 4);

    // 1. modulation vectors
    modulation_kernel<<<N6 / 256, 256, 0, stream>>>(vec, img_mod_w, img_mod_b, modI, D, N6);
    modulation_kernel<<<N6 / 256, 256, 0, stream>>>(vec, cond_mod_w, cond_mod_b, modC, D, N6);

    // 2. LN + modulate (shift1 @0, scale1 @D)
    ln_modulate_kernel<<<Li, 256, 0, stream>>>(img,  modI, 0, D, xmI, D);
    ln_modulate_kernel<<<Lc, 256, 0, stream>>>(cond, modC, 0, D, xmC, D);

    // 3. QKV GEMMs
    gemm_bf16_wmma<<<dim3(N3 / 128, Li / 128), 256, 0, stream>>>(
        xmI, img_qkv_w, img_qkv_b, qkvI, nullptr, Li, N3, D, 0);
    gemm_bf16_wmma<<<dim3(N3 / 128, Lc / 128), 256, 0, stream>>>(
        xmC, cond_qkv_w, cond_qkv_b, qkvC, nullptr, Lc, N3, D, 0);

    // 4. head split + rms norm (cond tokens first in concat order)
    qkv_rms_kernel<<<Li, 128, 0, stream>>>(qkvI, img_q_scale, img_k_scale, Qb, Kb, Vb, Lc, Lt);
    qkv_rms_kernel<<<Lc, 128, 0, stream>>>(qkvC, cond_q_scale, cond_k_scale, Qb, Kb, Vb, 0, Lt);

    // 5. flash attention over Lt=2304 tokens
    attn_kernel<<<dim3(Lt / 128, H), 256, 0, stream>>>(Qb, Kb, Vb, attnB, Lt);

    // 6. projection GEMMs (img rows start at token Lc in attnB)
    gemm_bf16_wmma<<<dim3(D / 128, Li / 128), 256, 0, stream>>>(
        attnB + (size_t)Lc * D, img_proj_w, img_proj_b, projI, nullptr, Li, D, D, 0);
    gemm_bf16_wmma<<<dim3(D / 128, Lc / 128), 256, 0, stream>>>(
        attnB, cond_proj_w, cond_proj_b, projC, nullptr, Lc, D, D, 0);

    // 7. first residual with gate1 (@2D)
    residual_gate_kernel<<<(int)(((long long)Li * D + 255) / 256), 256, 0, stream>>>(
        img, projI, modI, 2 * D, img1, D, (long long)Li * D);
    residual_gate_kernel<<<(int)(((long long)Lc * D + 255) / 256), 256, 0, stream>>>(
        cond, projC, modC, 2 * D, cond1, D, (long long)Lc * D);

    // 8. second LN + modulate (shift2 @3D, scale2 @4D)
    ln_modulate_kernel<<<Li, 256, 0, stream>>>(img1,  modI, 3 * D, 4 * D, xm2I, D);
    ln_modulate_kernel<<<Lc, 256, 0, stream>>>(cond1, modC, 3 * D, 4 * D, xm2C, D);

    // 9. MLP1 with GELU, bf16 out
    gemm_bf16_wmma<<<dim3(Dm / 128, Li / 128), 256, 0, stream>>>(
        xm2I, img_mlp_w1, img_mlp_b1, nullptr, hI, Li, Dm, D, 1);
    gemm_bf16_wmma<<<dim3(Dm / 128, Lc / 128), 256, 0, stream>>>(
        xm2C, cond_mlp_w1, cond_mlp_b1, nullptr, hC, Lc, Dm, D, 1);

    // 10. MLP2
    gemm_bf16_wmma<<<dim3(D / 128, Li / 128), 256, 0, stream>>>(
        hI, img_mlp_w2, img_mlp_b2, mlp2I, nullptr, Li, D, Dm, 0);
    gemm_bf16_wmma<<<dim3(D / 128, Lc / 128), 256, 0, stream>>>(
        hC, cond_mlp_w2, cond_mlp_b2, mlp2C, nullptr, Lc, D, Dm, 0);

    // 11. final residual with gate2 (@5D) -> d_out
    residual_gate_kernel<<<(int)(((long long)Li * D + 255) / 256), 256, 0, stream>>>(
        img1, mlp2I, modI, 5 * D, out_img, D, (long long)Li * D);
    residual_gate_kernel<<<(int)(((long long)Lc * D + 255) / 256), 256, 0, stream>>>(
        cond1, mlp2C, modC, 5 * D, out_cond, D, (long long)Lc * D);
}
